// encoder_BN_2_11046655885868
// MI455X (gfx1250) — compile-verified, matrix operand order checked by reading, not run on hardware
//
#include <hip/hip_runtime.h>
#include <hip/hip_bf16.h>

typedef __attribute__((ext_vector_type(16))) _Float16 v16h;
typedef __attribute__((ext_vector_type(8)))  _Float16 v8h;
typedef __attribute__((ext_vector_type(8)))  float    v8f;

#define EPSBN 1e-5f

// ---------------- transpose (B,3,N) -> (B,N,3) ----------------
__global__ void k_transpose(const float* __restrict__ xyz, float* __restrict__ pts, int B, int N) {
  int i = blockIdx.x * blockDim.x + threadIdx.x;
  if (i >= B * N) return;
  int b = i / N, n = i % N;
  pts[(size_t)(b * N + n) * 3 + 0] = xyz[(size_t)(b * 3 + 0) * N + n];
  pts[(size_t)(b * N + n) * 3 + 1] = xyz[(size_t)(b * 3 + 1) * N + n];
  pts[(size_t)(b * N + n) * 3 + 2] = xyz[(size_t)(b * 3 + 2) * N + n];
}

// ---------------- farthest point sampling: 1 block / batch ----------------
__global__ __launch_bounds__(512) void k_fps(const float* __restrict__ pts, int* __restrict__ out,
                                             int N, int npoint) {
  int b = blockIdx.x;
  const float* p = pts + (size_t)b * N * 3;
  __shared__ float sbest[512];
  __shared__ int   sbesti[512];
  __shared__ float cc[3];
  __shared__ int   sfar;
  int t = threadIdx.x;
  int per = N / 512;           // 8 for N=4096, 1 for N=512
  float dist[8];
  for (int j = 0; j < per; j++) dist[j] = 1e10f;
  if (t == 0) sfar = 0;
  __syncthreads();
  for (int it = 0; it < npoint; it++) {
    if (t == 0) {
      int f = sfar;
      out[b * npoint + it] = f;
      cc[0] = p[f * 3 + 0]; cc[1] = p[f * 3 + 1]; cc[2] = p[f * 3 + 2];
    }
    __syncthreads();
    float cx = cc[0], cy = cc[1], cz = cc[2];
    float best = -1.0f;
    int besti = t * per;
    for (int j = 0; j < per; j++) {
      int n = t * per + j;
      float dx = p[n * 3 + 0] - cx, dy = p[n * 3 + 1] - cy, dz = p[n * 3 + 2] - cz;
      float d = dx * dx + dy * dy + dz * dz;
      if (d < dist[j]) dist[j] = d;
      if (dist[j] > best) { best = dist[j]; besti = n; }
    }
    sbest[t] = best; sbesti[t] = besti;
    __syncthreads();
    for (int off = 256; off > 0; off >>= 1) {
      if (t < off) {
        if (sbest[t + off] > sbest[t]) { sbest[t] = sbest[t + off]; sbesti[t] = sbesti[t + off]; }
      }
      __syncthreads();
    }
    if (t == 0) sfar = sbesti[0];
    __syncthreads();
  }
}

// ---------------- gather 3d coords by index ----------------
__global__ void k_gather3(const float* __restrict__ pts, const int* __restrict__ idx,
                          float* __restrict__ out, int N, int S, int total) {
  int i = blockIdx.x * blockDim.x + threadIdx.x;
  if (i >= total) return;     // total = B*S
  int b = i / S, s = i % S;
  int n = idx[b * S + s];
  out[(size_t)i * 3 + 0] = pts[(size_t)(b * N + n) * 3 + 0];
  out[(size_t)i * 3 + 1] = pts[(size_t)(b * N + n) * 3 + 1];
  out[(size_t)i * 3 + 2] = pts[(size_t)(b * N + n) * 3 + 2];
}

// ---------------- ball query: first K indices within radius ----------------
__global__ void k_ball(const float* __restrict__ pts, const float* __restrict__ ctr,
                       int* __restrict__ out, int N, int S, int K, float r2, int total) {
  int i = blockIdx.x * blockDim.x + threadIdx.x;
  if (i >= total) return;     // total = B*S
  int b = i / S;
  const float* p = pts + (size_t)b * N * 3;
  float cx = ctr[(size_t)i * 3 + 0], cy = ctr[(size_t)i * 3 + 1], cz = ctr[(size_t)i * 3 + 2];
  int* o = out + (size_t)i * K;
  int cnt = 0, first = 0;
  for (int n = 0; n < N && cnt < K; n++) {
    float dx = p[n * 3 + 0] - cx, dy = p[n * 3 + 1] - cy, dz = p[n * 3 + 2] - cz;
    float d = dx * dx + dy * dy + dz * dz;
    if (!(d > r2)) {
      if (cnt == 0) first = n;
      o[cnt++] = n;
    }
  }
  for (int k = cnt; k < K; k++) o[k] = first;
}

// ---------------- build grouped A matrix (f16, K-padded): [feats(C), rel_xyz(3), pad] ----------------
__global__ void k_group(const float* __restrict__ pts, const float* __restrict__ ctr,
                        const float* __restrict__ feats, const int* __restrict__ idx,
                        _Float16* __restrict__ A, int N, int S, int K, int C, int Kp,
                        long long total) {
  long long i = (long long)blockIdx.x * blockDim.x + threadIdx.x;
  if (i >= total) return;     // total = B*S*K*Kp
  int c = (int)(i % Kp);
  long long row = i / Kp;
  int k = (int)(row % K);
  int s = (int)((row / K) % S);
  int b = (int)(row / ((long long)K * S));
  int n = idx[((size_t)(b * S + s)) * K + k];
  if (n >= N) n = N - 1;
  float v = 0.0f;
  if (c < C) {
    v = feats[((size_t)b * N + n) * C + c];
  } else if (c < C + 3) {
    int d = c - C;
    v = pts[((size_t)b * N + n) * 3 + d] - ctr[((size_t)(b * S + s)) * 3 + d];
  }
  A[i] = (_Float16)v;
}

// ---------------- SA3 group-all: [xyz(3), feats(C), pad] ----------------
__global__ void k_group_sa3(const float* __restrict__ l2xyz, const float* __restrict__ l2feat,
                            _Float16* __restrict__ A, int S, int C, int Kp, long long total) {
  long long i = (long long)blockIdx.x * blockDim.x + threadIdx.x;
  if (i >= total) return;     // total = B*S*Kp
  int c = (int)(i % Kp);
  long long row = i / Kp;     // row = b*S + s
  float v = 0.0f;
  if (c < 3) v = l2xyz[(size_t)row * 3 + c];
  else if (c < 3 + C) v = l2feat[(size_t)row * C + (c - 3)];
  A[i] = (_Float16)v;
}

// ---------------- pad/convert weights: (cout,cin) f32 -> (cout,Kp) f16 ----------------
__global__ void k_wpad(const float* __restrict__ W, _Float16* __restrict__ Wp,
                       int cout, int cin, int Kp) {
  int i = blockIdx.x * blockDim.x + threadIdx.x;
  if (i >= cout * Kp) return;
  int n = i / Kp, k = i % Kp;
  Wp[i] = (k < cin) ? (_Float16)W[(size_t)n * cin + k] : (_Float16)0.0f;
}

// ---------------- WMMA GEMM, N-blocked (NB sub-tiles of 16 cols per wave) ----------------
// Y(MxN,f32) = A(MxKp,f16) * Wp(NxKp,f16)^T
// tile = tg*(M/16) + tm : all 8 waves of a block share column-group tg, so the
// B k-slice (NB*16 rows x 32 k, <=8KB) is staged cooperatively into LDS via
// global_load_async_to_lds_b128 (+ s_wait_asynccnt) and consumed as ds_load_b128.
template <int NB>
__global__ __launch_bounds__(256) void k_gemm_t(const _Float16* __restrict__ A,
                                                const _Float16* __restrict__ Wp,
                                                float* __restrict__ Y,
                                                int M, int N, int Kp, int tiles) {
  __shared__ _Float16 sB[NB * 16 * 32];
  int wave = threadIdx.x >> 5;
  int lane = threadIdx.x & 31;
  int mt = M >> 4;
  int tile = blockIdx.x * 8 + wave;
  if (tile >= tiles) tile = tiles - 1;           // never straddles: tiles % 8 == 0
  int tm = tile % mt;
  int tg  = tile / mt;
  int tgb = (blockIdx.x * 8) / mt;               // block-uniform (== tg)
  int half = lane >> 4;                          // 0: K 0-7/16-23, 1: K 8-15/24-31
  int l = lane & 15;
  const _Float16* arow = A + (size_t)(tm * 16 + l) * Kp + half * 8;

  v8f acc[NB];
#pragma unroll
  for (int j = 0; j < NB; j++)
#pragma unroll
    for (int q = 0; q < 8; q++) acc[j][q] = 0.0f;

  for (int k0 = 0; k0 < Kp; k0 += 32) {
    // ---- async stage of B slice: rows [tgb*NB*16, +NB*16), k [k0, k0+32) ----
    for (int g = threadIdx.x; g < NB * 64; g += 256) {
      int row = g >> 2;
      int ko  = (g & 3) * 8;                     // 4 x 16B granules per 32-f16 row
      const _Float16* src = Wp + (size_t)(tgb * NB * 16 + row) * Kp + k0 + ko;
      unsigned loff = (unsigned)(size_t)(&sB[row * 32 + ko]);  // low 32b of flat LDS ptr = LDS addr
      asm volatile("global_load_async_to_lds_b128 %0, %1, off"
                   :: "v"(loff), "v"(src) : "memory");
    }
    asm volatile("s_wait_asynccnt 0" ::: "memory");
    __syncthreads();

    if (k0 + 32 < Kp) __builtin_prefetch(arow + k0 + 64, 0, 1);   // global_prefetch_b8
    v8h a0 = *(const v8h*)(arow + k0);
    v8h a1 = *(const v8h*)(arow + k0 + 16);
    v16h a = __builtin_shufflevector(a0, a1, 0,1,2,3,4,5,6,7,8,9,10,11,12,13,14,15);

#pragma unroll
    for (int j = 0; j < NB; j++) {
      const _Float16* brow = &sB[(j * 16 + l) * 32] + half * 8;
      v8h b0 = *(const v8h*)(brow);
      v8h b1 = *(const v8h*)(brow + 16);
      v16h b = __builtin_shufflevector(b0, b1, 0,1,2,3,4,5,6,7,8,9,10,11,12,13,14,15);
      acc[j] = __builtin_amdgcn_wmma_f32_16x16x32_f16(false, a, false, b, (short)0, acc[j],
                                                      false, false);
    }
    __syncthreads();                             // before next slice overwrites sB
  }

  // D layout: VGPR q -> row (q + 8*half), col = lane&15
#pragma unroll
  for (int j = 0; j < NB; j++) {
    float* yb = Y + (size_t)(tm * 16 + half * 8) * N + (tg * NB + j) * 16 + l;
#pragma unroll
    for (int q = 0; q < 8; q++) yb[(size_t)q * N] = acc[j][q];
  }
}

// ---------------- zero f32 buffer ----------------
__global__ void k_zero(float* p, int n) {
  int i = blockIdx.x * blockDim.x + threadIdx.x;
  if (i < n) p[i] = 0.0f;
}

// ---------------- per-channel sum / sumsq, linear coalesced sweep ----------------
__global__ __launch_bounds__(256) void k_stats(const float* __restrict__ Y,
                                               float* __restrict__ sum, float* __restrict__ sumsq,
                                               int M, int N, long long epb) {
  long long total = (long long)M * N;
  long long base = (long long)blockIdx.x * epb;
  long long end = base + epb; if (end > total) end = total;
  if (N <= 256 && (256 % N) == 0) {
    // each thread owns one fixed channel: register accumulate, one atomic at end
    float s = 0.0f, s2 = 0.0f;
    int c = (int)((base + threadIdx.x) % N);
    for (long long i = base + threadIdx.x; i < end; i += 256) {
      float v = Y[i]; s += v; s2 += v * v;
    }
    atomicAdd(&sum[c], s); atomicAdd(&sumsq[c], s2);
  } else {
    __shared__ float ls[2048];                 // N <= 1024: [sum(N) | sumsq(N)]
    for (int i = threadIdx.x; i < 2 * N; i += 256) ls[i] = 0.0f;
    __syncthreads();
    for (long long i = base + threadIdx.x; i < end; i += 256) {
      float v = Y[i]; int c = (int)(i % N);
      atomicAdd(&ls[c], v); atomicAdd(&ls[N + c], v * v);
    }
    __syncthreads();
    for (int i = threadIdx.x; i < N; i += 256) {
      atomicAdd(&sum[i], ls[i]); atomicAdd(&sumsq[i], ls[N + i]);
    }
  }
}

// ---------------- BN(+ReLU) -> next-layer f16 A (K-padded) ----------------
__global__ void k_bnrelu_f16(const float* __restrict__ Y, const float* __restrict__ sum,
                             const float* __restrict__ sumsq, const float* __restrict__ gamma,
                             const float* __restrict__ beta, _Float16* __restrict__ A,
                             int M, int N, int Kp2, long long total) {
  long long i = (long long)blockIdx.x * blockDim.x + threadIdx.x;
  if (i >= total) return;     // total = M*Kp2
  int c = (int)(i % Kp2);
  long long row = i / Kp2;
  if (c >= N) { A[i] = (_Float16)0.0f; return; }
  float mean = sum[c] / (float)M;
  float var  = sumsq[c] / (float)M - mean * mean;
  float v = gamma[c] * (Y[(size_t)row * N + c] - mean) * rsqrtf(var + EPSBN) + beta[c];
  A[i] = (_Float16)(v > 0.0f ? v : 0.0f);
}

// ---------------- BN(+ReLU) in place on f32 Y (last layer of a scale) ----------------
__global__ void k_bnrelu_inplace(float* __restrict__ Y, const float* __restrict__ sum,
                                 const float* __restrict__ sumsq, const float* __restrict__ gamma,
                                 const float* __restrict__ beta, int M, int N, long long total) {
  long long i = (long long)blockIdx.x * blockDim.x + threadIdx.x;
  if (i >= total) return;     // total = M*N
  int c = (int)(i % N);
  float mean = sum[c] / (float)M;
  float var  = sumsq[c] / (float)M - mean * mean;
  float v = gamma[c] * (Y[i] - mean) * rsqrtf(var + EPSBN) + beta[c];
  Y[i] = v > 0.0f ? v : 0.0f;
}

// ---------------- max over K samples -> feature buffer slice ----------------
__global__ void k_maxpool(const float* __restrict__ Y, float* __restrict__ out,
                          int BS, int K, int N, int ldout, int choff) {
  int i = blockIdx.x * blockDim.x + threadIdx.x;
  if (i >= BS * N) return;
  int bs = i / N, ch = i % N;
  float m = -3.4e38f;
  for (int k = 0; k < K; k++) m = fmaxf(m, Y[((size_t)bs * K + k) * N + ch]);
  out[(size_t)bs * ldout + choff + ch] = m;
}

// ---------------- FC + BN(axis 0, batch=8) + ReLU ----------------
__global__ void k_fc(const float* __restrict__ X, const float* __restrict__ W,
                     const float* __restrict__ gamma, const float* __restrict__ beta,
                     float* __restrict__ out, int Kd, int N) {
  int ch = blockIdx.x * blockDim.x + threadIdx.x;
  if (ch >= N) return;
  float y[8];
  const float* w = W + (size_t)ch * Kd;
  for (int b = 0; b < 8; b++) {
    const float* x = X + (size_t)b * Kd;
    float s = 0.0f;
    for (int k = 0; k < Kd; k++) s += x[k] * w[k];
    y[b] = s;
  }
  float m = 0.0f;
  for (int b = 0; b < 8; b++) m += y[b];
  m *= 0.125f;
  float v = 0.0f;
  for (int b = 0; b < 8; b++) { float d = y[b] - m; v += d * d; }
  v *= 0.125f;
  float rs = rsqrtf(v + EPSBN);
  for (int b = 0; b < 8; b++) {
    float t = gamma[ch] * (y[b] - m) * rs + beta[ch];
    out[(size_t)b * N + ch] = t > 0.0f ? t : 0.0f;
  }
}

// =====================================================================

extern "C" void kernel_launch(void* const* d_in, const int* in_sizes, int n_in,
                              void* d_out, int out_size, void* d_ws, size_t ws_size,
                              hipStream_t stream) {
  (void)in_sizes; (void)n_in; (void)out_size; (void)ws_size;
  const int B = 8, N0 = 4096;

  // ---- workspace carve (bump allocator, 256B aligned) ----
  size_t off = 0;
  auto alloc = [&](size_t sz) -> char* {
    char* p = (char*)d_ws + off;
    off = (off + sz + 255) & ~(size_t)255;
    return p;
  };
  float*    pts     = (float*)   alloc((size_t)B * N0 * 3 * 4);
  float*    l1_xyz  = (float*)   alloc((size_t)B * 512 * 3 * 4);
  float*    l2_xyz  = (float*)   alloc((size_t)B * 128 * 3 * 4);
  int*      fpsidx  = (int*)     alloc((size_t)B * 512 * 4);
  int*      ballidx = (int*)     alloc((size_t)B * 512 * 128 * 4);
  float*    l1_feat = (float*)   alloc((size_t)B * 512 * 320 * 4);
  float*    l2_feat = (float*)   alloc((size_t)B * 128 * 640 * 4);
  float*    l3_feat = (float*)   alloc((size_t)B * 1024 * 4);
  float*    fc1buf  = (float*)   alloc((size_t)B * 512 * 4);
  float*    fc2buf  = (float*)   alloc((size_t)B * 256 * 4);
  float*    stats   = (float*)   alloc(2 * 1024 * 4);          // sum | sumsq (offset 1024)
  _Float16* Wp      = (_Float16*)alloc((size_t)1024 * 512 * 2);
  _Float16* Abuf    = (_Float16*)alloc((size_t)524288 * 96 * 2);
  float*    Ybuf    = (float*)   alloc((size_t)524288 * 128 * 4);

  auto pad32 = [](int c) { return (c + 31) & ~31; };

  auto launch_gemm = [&](int M, int N, int Kp) {
    int ntx = N >> 4;
    int NB = (ntx >= 8) ? 8 : ntx;       // 2,4,6,8
    int G = ntx / NB;
    int tiles = (M >> 4) * G;
    int blocks = (tiles + 7) / 8;
    switch (NB) {
      case 2: k_gemm_t<2><<<blocks, 256, 0, stream>>>(Abuf, Wp, Ybuf, M, N, Kp, tiles); break;
      case 4: k_gemm_t<4><<<blocks, 256, 0, stream>>>(Abuf, Wp, Ybuf, M, N, Kp, tiles); break;
      case 6: k_gemm_t<6><<<blocks, 256, 0, stream>>>(Abuf, Wp, Ybuf, M, N, Kp, tiles); break;
      default: k_gemm_t<8><<<blocks, 256, 0, stream>>>(Abuf, Wp, Ybuf, M, N, Kp, tiles); break;
    }
  };

  // conv layer: A(M x pad32(cin), f16) -> Y(M x cout, f32); BN(+ReLU); optionally emit next f16 A
  auto run_conv = [&](int j, int M, int cin, int cout, bool last) {
    const float* W     = (const float*)d_in[1 + 4 * j];
    const float* gamma = (const float*)d_in[3 + 4 * j];
    const float* beta  = (const float*)d_in[4 + 4 * j];
    int Kp = pad32(cin);
    int nw = cout * Kp;
    k_wpad<<<(nw + 255) / 256, 256, 0, stream>>>(W, Wp, cout, cin, Kp);
    launch_gemm(M, cout, Kp);
    k_zero<<<(2 * 1024 + 255) / 256, 256, 0, stream>>>(stats, 2 * 1024);
    long long elems = (long long)M * cout;
    long long epb = 1 << 18;
    int sblocks = (int)((elems + epb - 1) / epb);
    k_stats<<<sblocks, 256, 0, stream>>>(Ybuf, stats, stats + 1024, M, cout, epb);
    if (!last) {
      int Kp2 = pad32(cout);
      long long tot = (long long)M * Kp2;
      k_bnrelu_f16<<<(unsigned)((tot + 255) / 256), 256, 0, stream>>>(
          Ybuf, stats, stats + 1024, gamma, beta, Abuf, M, cout, Kp2, tot);
    } else {
      long long tot = (long long)M * cout;
      k_bnrelu_inplace<<<(unsigned)((tot + 255) / 256), 256, 0, stream>>>(
          Ybuf, stats, stats + 1024, gamma, beta, M, cout, tot);
    }
  };

  // ---- input prep ----
  const float* xyz = (const float*)d_in[0];
  k_transpose<<<(B * N0 + 255) / 256, 256, 0, stream>>>(xyz, pts, B, N0);

  // ================= SA1 (N=4096 -> S=512, feats: none) =================
  k_fps<<<B, 512, 0, stream>>>(pts, fpsidx, N0, 512);
  k_gather3<<<(B * 512 + 255) / 256, 256, 0, stream>>>(pts, fpsidx, l1_xyz, N0, 512, B * 512);
  {
    const int   Ks[3] = {16, 32, 128};
    const float rs[3] = {0.1f, 0.2f, 0.4f};
    const int   ch[3][4] = {{3, 32, 32, 64}, {3, 64, 64, 128}, {3, 64, 96, 128}};
    int choff = 0;
    for (int sc = 0; sc < 3; sc++) {
      int K = Ks[sc], S = 512, M = B * S * K;
      k_ball<<<(B * S + 255) / 256, 256, 0, stream>>>(pts, l1_xyz, ballidx, N0, S, K,
                                                      rs[sc] * rs[sc], B * S);
      long long tot = (long long)M * 32;
      k_group<<<(unsigned)((tot + 255) / 256), 256, 0, stream>>>(
          pts, l1_xyz, (const float*)nullptr, ballidx, Abuf, N0, S, K, 0, 32, tot);
      for (int L = 0; L < 3; L++) run_conv(sc * 3 + L, M, ch[sc][L], ch[sc][L + 1], L == 2);
      k_maxpool<<<(B * S * ch[sc][3] + 255) / 256, 256, 0, stream>>>(
          Ybuf, l1_feat, B * S, K, ch[sc][3], 320, choff);
      choff += ch[sc][3];
    }
  }

  // ================= SA2 (N=512 -> S=128, feats: 320ch) =================
  k_fps<<<B, 512, 0, stream>>>(l1_xyz, fpsidx, 512, 128);
  k_gather3<<<(B * 128 + 255) / 256, 256, 0, stream>>>(l1_xyz, fpsidx, l2_xyz, 512, 128, B * 128);
  {
    const int   Ks[3] = {32, 64, 128};
    const float rs[3] = {0.2f, 0.4f, 0.8f};
    const int   ch[3][4] = {{323, 64, 64, 128}, {323, 128, 128, 256}, {323, 128, 128, 256}};
    int choff = 0;
    for (int sc = 0; sc < 3; sc++) {
      int K = Ks[sc], S = 128, M = B * S * K;
      k_ball<<<(B * S + 255) / 256, 256, 0, stream>>>(l1_xyz, l2_xyz, ballidx, 512, S, K,
                                                      rs[sc] * rs[sc], B * S);
      int Kp = pad32(323);  // 352
      long long tot = (long long)M * Kp;
      k_group<<<(unsigned)((tot + 255) / 256), 256, 0, stream>>>(
          l1_xyz, l2_xyz, l1_feat, ballidx, Abuf, 512, S, K, 320, Kp, tot);
      for (int L = 0; L < 3; L++) run_conv(9 + sc * 3 + L, M, ch[sc][L], ch[sc][L + 1], L == 2);
      k_maxpool<<<(B * S * ch[sc][3] + 255) / 256, 256, 0, stream>>>(
          Ybuf, l2_feat, B * S, K, ch[sc][3], 640, choff);
      choff += ch[sc][3];
    }
  }

  // ================= SA3 (group all: [xyz(3), feats(640)] -> 1024ch) =================
  {
    int S = 128, M = B * S;       // 1024 rows
    int Kp = pad32(643);          // 672
    long long tot = (long long)M * Kp;
    k_group_sa3<<<(unsigned)((tot + 255) / 256), 256, 0, stream>>>(l2_xyz, l2_feat, Abuf, S, 640, Kp, tot);
    const int ch[4] = {643, 256, 512, 1024};
    for (int L = 0; L < 3; L++) run_conv(18 + L, M, ch[L], ch[L + 1], L == 2);
    k_maxpool<<<(B * 1024 + 255) / 256, 256, 0, stream>>>(Ybuf, l3_feat, B, S, 1024, 1024, 0);
  }

  // ================= FC head =================
  k_fc<<<(512 + 255) / 256, 256, 0, stream>>>(l3_feat, (const float*)d_in[1 + 4 * 21],
                                              (const float*)d_in[3 + 4 * 21],
                                              (const float*)d_in[4 + 4 * 21], fc1buf, 1024, 512);
  k_fc<<<(256 + 255) / 256, 256, 0, stream>>>(fc1buf, (const float*)d_in[1 + 4 * 22],
                                              (const float*)d_in[3 + 4 * 22],
                                              (const float*)d_in[4 + 4 * 22], fc2buf, 512, 256);
  k_fc<<<(128 + 255) / 256, 256, 0, stream>>>(fc2buf, (const float*)d_in[1 + 4 * 23],
                                              (const float*)d_in[3 + 4 * 23],
                                              (const float*)d_in[4 + 4 * 23], (float*)d_out, 256, 128);
}
